// Block_29291676958810
// MI455X (gfx1250) — matrix operand order repaired and checked by reading, not compiled, not run on hardware
//
#include <hip/hip_runtime.h>
#include <hip/hip_bf16.h>

typedef __attribute__((ext_vector_type(16))) __bf16 v16bf;
typedef __attribute__((ext_vector_type(8)))  __bf16 v8bf;
typedef __attribute__((ext_vector_type(8)))  float  v8f;

#define B_   2
#define T_   2048
#define C_   1024
#define H_   16
#define DH_  64
#define DFF_ 4096
#define BT_  (B_ * T_)

// --------------------------------------------------------------------------
// Async global->LDS copy (CDNA5 GLOBAL_LOAD_ASYNC_TO_LDS_B128, ASYNCcnt).
// The clang builtin's prototype differs across toolchains, so use inline asm
// (GV addressing: vdst = per-lane LDS byte address, vaddr = 64-bit global).
// --------------------------------------------------------------------------
#if defined(__has_builtin)
#  if __has_builtin(__builtin_amdgcn_s_wait_asynccnt)
#    define HAVE_WAIT_ASYNC_BUILTIN 1
#  endif
#endif
#ifndef HAVE_WAIT_ASYNC_BUILTIN
#  define HAVE_WAIT_ASYNC_BUILTIN 0
#endif

__device__ __forceinline__ void async_copy16(const void* g, void* l) {
    // flat LDS address truncates to the 32-bit LDS offset (aperture mapping)
    unsigned lds_off = (unsigned)(size_t)l;
    unsigned long long ga = (unsigned long long)(size_t)g;
    asm volatile("global_load_async_to_lds_b128 %0, %1, off"
                 :
                 : "v"(lds_off), "v"(ga)
                 : "memory");
}
__device__ __forceinline__ void async_wait0() {
#if HAVE_WAIT_ASYNC_BUILTIN
    __builtin_amdgcn_s_wait_asynccnt(0);
#else
    asm volatile("s_wait_asynccnt 0x0" ::: "memory");
#endif
}

// ---------------------------------------------------------------------------
// f32 -> bf16 conversion (grid-stride)
// ---------------------------------------------------------------------------
__global__ void cvt_f32_bf16(const float* __restrict__ in, __bf16* __restrict__ out, long long n) {
    long long i = (long long)blockIdx.x * blockDim.x + threadIdx.x;
    long long stride = (long long)gridDim.x * blockDim.x;
    for (; i < n; i += stride) out[i] = (__bf16)in[i];
}

// ---------------------------------------------------------------------------
// Tiled bf16 GEMM, f32 accumulate via v_wmma_f32_16x16x32_bf16.
// BM=128, BN=64, BK=32. 256 threads = 8 waves; wave (wm 0..3, wn 0..1)
// computes a 32x32 quadrant = 2x2 WMMA tiles. A tile staged to LDS with
// async-to-LDS copies (ASYNCcnt); B tile staged transposed synchronously.
// MODE 0: out bf16 at q[((b*H+bx)*T+t)*DH + n]          (Q/K projection)
// MODE 1: out bf16 at v[((b*H+bx)*DH + n)*T + t]        (V, transposed)
// MODE 2: outF[row*ldo+col] = acc + bias[col] + resid[row*ldo+col]
// MODE 3: outB[row*ldo+col] = silu(acc + bias[col])     (FFN1)
// ---------------------------------------------------------------------------
template <int MODE>
__global__ __launch_bounds__(256) void gemm_bf16(
    const __bf16* __restrict__ A,  int lda,
    const __bf16* __restrict__ Bm, int ldb,
    const float*  __restrict__ bias,
    const float*  __restrict__ resid,
    float*        __restrict__ outF,
    __bf16*       __restrict__ outB,
    int K, int ldo)
{
    constexpr int BM = 128, BN = 64, BK = 32, PIT = 40;
    __shared__ __align__(16) __bf16 As[BM * PIT];
    __shared__ __align__(16) __bf16 Bs[BN * PIT];

    const int tid  = threadIdx.x;
    const int lane = tid & 31;
    const int wave = tid >> 5;
    const int wm   = wave & 3;
    const int wn   = wave >> 2;
    const int row0 = blockIdx.y * BM;

    const long long bbase = (MODE <= 1) ? (long long)blockIdx.x * (C_ * DH_) : 0;
    const int bcol0       = (MODE <= 1) ? 0 : blockIdx.x * BN;

    const int mlane = lane & 15;
    const int kbA   = (lane < 16) ? 0 : 8;    // A-frag K split per ISA layout
    const int kbB   = (lane < 16) ? 0 : 8;    // B-frag K split per ISA layout
    const int rowh  = (lane < 16) ? 0 : 8;    // C/D row split per ISA layout

    v8f acc[2][2] = {};

    for (int k0 = 0; k0 < K; k0 += BK) {
        __syncthreads();   // previous iteration's LDS reads complete

        // A tile 128x32 -> LDS (row-major, pitch 40) via async-to-LDS
        #pragma unroll
        for (int c = 0; c < 2; ++c) {
            int chunk = tid + 256 * c;            // 512 chunks of 8 bf16
            int r  = chunk >> 2;
            int kc = (chunk & 3) * 8;
            async_copy16(&A[(long long)(row0 + r) * lda + k0 + kc],
                         &As[r * PIT + kc]);
        }
        // B tile 32x64 -> LDS transposed Bs[n][k] (sync: transpose in flight)
        {
            int kk = tid >> 3;
            int nc = (tid & 7) * 8;
            v8bf tmp = *(const v8bf*)&Bm[bbase + (long long)(k0 + kk) * ldb + bcol0 + nc];
            #pragma unroll
            for (int j = 0; j < 8; ++j) Bs[(nc + j) * PIT + kk] = tmp[j];
        }
        // warm L2 for the next K-step while this tile is in flight
        if (k0 + BK < K) {
            __builtin_prefetch(&A[(long long)(row0 + (tid >> 1)) * lda + k0 + BK], 0, 1);
            __builtin_prefetch(&Bm[bbase + (long long)(k0 + BK + (tid >> 3)) * ldb + bcol0], 0, 1);
        }
        async_wait0();     // this wave's async-to-LDS copies landed
        __syncthreads();   // all waves' tile writes visible

        v16bf af[2], bfrag[2];
        #pragma unroll
        for (int i = 0; i < 2; ++i) {
            int m = wm * 32 + i * 16 + mlane;
            v8bf lo = *(const v8bf*)&As[m * PIT + kbA];
            v8bf hi = *(const v8bf*)&As[m * PIT + kbA + 16];
            #pragma unroll
            for (int j = 0; j < 8; ++j) { af[i][j] = lo[j]; af[i][j + 8] = hi[j]; }
        }
        #pragma unroll
        for (int i = 0; i < 2; ++i) {
            int n = wn * 32 + i * 16 + mlane;
            v8bf lo = *(const v8bf*)&Bs[n * PIT + kbB];
            v8bf hi = *(const v8bf*)&Bs[n * PIT + kbB + 16];
            #pragma unroll
            for (int j = 0; j < 8; ++j) { bfrag[i][j] = lo[j]; bfrag[i][j + 8] = hi[j]; }
        }
        #pragma unroll
        for (int i = 0; i < 2; ++i)
            #pragma unroll
            for (int j = 0; j < 2; ++j)
                acc[i][j] = __builtin_amdgcn_wmma_f32_16x16x32_bf16(
                    false, af[i], false, bfrag[j], (short)0, acc[i][j], false, false);
    }

    // Epilogue (C/D layout: lane = col, VGPR r = row r + rowh)
    #pragma unroll
    for (int i = 0; i < 2; ++i) {
        #pragma unroll
        for (int j = 0; j < 2; ++j) {
            #pragma unroll
            for (int r = 0; r < 8; ++r) {
                int grow = row0 + wm * 32 + i * 16 + r + rowh;
                int lcol = wn * 32 + j * 16 + mlane;
                float v = acc[i][j][r];
                if (MODE == 0) {
                    int b = grow >> 11, t = grow & (T_ - 1);
                    outB[(((long long)(b * H_ + blockIdx.x)) * T_ + t) * DH_ + lcol] = (__bf16)v;
                } else if (MODE == 1) {
                    int b = grow >> 11, t = grow & (T_ - 1);
                    outB[(((long long)(b * H_ + blockIdx.x)) * DH_ + lcol) * T_ + t] = (__bf16)v;
                } else if (MODE == 2) {
                    int col = bcol0 + lcol;
                    outF[(long long)grow * ldo + col] =
                        v + bias[col] + resid[(long long)grow * ldo + col];
                } else {
                    int col = bcol0 + lcol;
                    float xv = v + bias[col];
                    float sv = xv / (1.0f + __expf(-xv));
                    outB[(long long)grow * ldo + col] = (__bf16)sv;
                }
            }
        }
    }
}

// ---------------------------------------------------------------------------
// Flash attention, causal, scale = C^-0.5 = 1/32.
// Grid: B*H*(T/64) blocks of 128 threads; each wave owns one 16-row q tile.
// Scores: Q(16x64) x K^T via 2 WMMAs/16-key subtile. Online softmax with
// 16-lane shuffle row reductions. P re-laid-out via wave-private LDS tile,
// then P(16x32) x V(32x64) via 4 WMMAs (V stored transposed [B,H,DH,T]).
// ---------------------------------------------------------------------------
__global__ __launch_bounds__(128) void attn_kernel(
    const __bf16* __restrict__ q, const __bf16* __restrict__ k,
    const __bf16* __restrict__ vT, __bf16* __restrict__ out)
{
    __shared__ __align__(16) __bf16 Pt[4][16 * 40];
    const int tid  = threadIdx.x;
    const int lane = tid & 31;
    const int wave = tid >> 5;
    const int bh   = blockIdx.x >> 5;                       // 0..B*H-1
    const int t0   = ((blockIdx.x & 31) * 4 + wave) * 16;   // query tile base
    const int b    = bh >> 4;
    const int h    = bh & (H_ - 1);

    const __bf16* qh = q  + (long long)bh * T_ * DH_;
    const __bf16* kh = k  + (long long)bh * T_ * DH_;
    const __bf16* vh = vT + (long long)bh * DH_ * T_;
    __bf16* Pw = Pt[wave];

    const int mlane = lane & 15;
    const int kbA   = (lane < 16) ? 0 : 8;
    const int kbB   = (lane < 16) ? 0 : 8;
    const int rowh  = (lane < 16) ? 0 : 8;
    const float scale = 0.03125f;

    // Q A-fragments for head-dim 0..31 and 32..63 (loaded once)
    v16bf aq[2];
    #pragma unroll
    for (int f = 0; f < 2; ++f) {
        const __bf16* base = &qh[(long long)(t0 + mlane) * DH_ + f * 32 + kbA];
        v8bf lo = *(const v8bf*)base;
        v8bf hi = *(const v8bf*)(base + 16);
        #pragma unroll
        for (int j = 0; j < 8; ++j) { aq[f][j] = lo[j]; aq[f][j + 8] = hi[j]; }
    }

    v8f o[4] = {};
    float mrun[8], lrun[8];
    #pragma unroll
    for (int r = 0; r < 8; ++r) { mrun[r] = -1e30f; lrun[r] = 0.0f; }

    for (int s0 = 0; s0 < t0 + 16; s0 += 32) {
        float p[2][8];
        float mcur[8];
        #pragma unroll
        for (int r = 0; r < 8; ++r) mcur[r] = -1e30f;

        #pragma unroll
        for (int sub = 0; sub < 2; ++sub) {
            const int ss = s0 + sub * 16;
            v8f sacc = {};
            #pragma unroll
            for (int f = 0; f < 2; ++f) {
                const __bf16* base = &kh[(long long)(ss + mlane) * DH_ + f * 32 + kbB];
                v8bf lo = *(const v8bf*)base;
                v8bf hi = *(const v8bf*)(base + 16);
                v16bf bk;
                #pragma unroll
                for (int j = 0; j < 8; ++j) { bk[j] = lo[j]; bk[j + 8] = hi[j]; }
                sacc = __builtin_amdgcn_wmma_f32_16x16x32_bf16(
                    false, aq[f], false, bk, (short)0, sacc, false, false);
            }
            #pragma unroll
            for (int r = 0; r < 8; ++r) {
                int trow = t0 + r + rowh;
                int scol = ss + mlane;
                float val = (scol <= trow) ? sacc[r] * scale : -1e30f;
                p[sub][r] = val;
                mcur[r] = fmaxf(mcur[r], val);
            }
        }
        // row max across the 16 lanes holding this row's columns
        #pragma unroll
        for (int mask = 1; mask < 16; mask <<= 1)
            #pragma unroll
            for (int r = 0; r < 8; ++r)
                mcur[r] = fmaxf(mcur[r], __shfl_xor(mcur[r], mask, 32));

        float alpha[8];
        #pragma unroll
        for (int r = 0; r < 8; ++r) {
            float mnew = fmaxf(mrun[r], mcur[r]);
            alpha[r] = __expf(mrun[r] - mnew);
            mrun[r] = mnew;
        }
        float lsum[8];
        #pragma unroll
        for (int r = 0; r < 8; ++r) {
            p[0][r] = __expf(p[0][r] - mrun[r]);
            p[1][r] = __expf(p[1][r] - mrun[r]);
            lsum[r] = p[0][r] + p[1][r];
        }
        #pragma unroll
        for (int mask = 1; mask < 16; mask <<= 1)
            #pragma unroll
            for (int r = 0; r < 8; ++r)
                lsum[r] += __shfl_xor(lsum[r], mask, 32);
        #pragma unroll
        for (int r = 0; r < 8; ++r) lrun[r] = lrun[r] * alpha[r] + lsum[r];

        // rescale running output
        #pragma unroll
        for (int f = 0; f < 4; ++f)
            #pragma unroll
            for (int r = 0; r < 8; ++r) o[f][r] *= alpha[r];

        // P (C-layout) -> LDS -> A-fragment layout (wave-private tile,
        // LDS ops are in-order within a wave)
        #pragma unroll
        for (int sub = 0; sub < 2; ++sub)
            #pragma unroll
            for (int r = 0; r < 8; ++r)
                Pw[(r + rowh) * 40 + sub * 16 + mlane] = (__bf16)p[sub][r];

        v16bf ap;
        {
            v8bf lo = *(const v8bf*)&Pw[mlane * 40 + kbA];
            v8bf hi = *(const v8bf*)&Pw[mlane * 40 + kbA + 16];
            #pragma unroll
            for (int j = 0; j < 8; ++j) { ap[j] = lo[j]; ap[j + 8] = hi[j]; }
        }
        // P(16x32) x V(32x64): V^T rows are contiguous in keys
        #pragma unroll
        for (int f = 0; f < 4; ++f) {
            const __bf16* base = &vh[(long long)(f * 16 + mlane) * T_ + s0 + kbB];
            v8bf lo = *(const v8bf*)base;
            v8bf hi = *(const v8bf*)(base + 16);
            v16bf bv;
            #pragma unroll
            for (int j = 0; j < 8; ++j) { bv[j] = lo[j]; bv[j + 8] = hi[j]; }
            o[f] = __builtin_amdgcn_wmma_f32_16x16x32_bf16(
                false, ap, false, bv, (short)0, o[f], false, false);
        }
    }

    // write attn (concat heads) as bf16 [B,T,C] for the Wo GEMM
    #pragma unroll
    for (int f = 0; f < 4; ++f)
        #pragma unroll
        for (int r = 0; r < 8; ++r) {
            int trow = t0 + r + rowh;
            out[((long long)(b * T_ + trow)) * C_ + h * DH_ + f * 16 + mlane] =
                (__bf16)(o[f][r] / lrun[r]);
        }
}

// ---------------------------------------------------------------------------
// RMSNorm over C=1024: one block per row; writes f32 (residual) + bf16 (GEMM)
// ---------------------------------------------------------------------------
__global__ __launch_bounds__(256) void rmsnorm_kernel(
    const float* __restrict__ in, const float* __restrict__ g,
    float* __restrict__ outF, __bf16* __restrict__ outB)
{
    __shared__ float red[8];
    const long long row = blockIdx.x;
    const float* x = in + row * C_;
    float ss = 0.0f;
    #pragma unroll
    for (int i = 0; i < 4; ++i) {
        float v = x[threadIdx.x + 256 * i];
        ss += v * v;
    }
    #pragma unroll
    for (int m = 1; m < 32; m <<= 1) ss += __shfl_xor(ss, m, 32);
    if ((threadIdx.x & 31) == 0) red[threadIdx.x >> 5] = ss;
    __syncthreads();
    float tot = 0.0f;
    #pragma unroll
    for (int w = 0; w < 8; ++w) tot += red[w];
    const float rinv = rsqrtf(tot * (1.0f / (float)C_) + 1e-6f);
    #pragma unroll
    for (int i = 0; i < 4; ++i) {
        int c = threadIdx.x + 256 * i;
        float v = x[c] * rinv * g[c];
        if (outF) outF[row * C_ + c] = v;
        if (outB) outB[row * C_ + c] = (__bf16)v;
    }
}

// ---------------------------------------------------------------------------
extern "C" void kernel_launch(void* const* d_in, const int* in_sizes, int n_in,
                              void* d_out, int out_size, void* d_ws, size_t ws_size,
                              hipStream_t stream) {
    const float* x  = (const float*)d_in[0];
    const float* Wq = (const float*)d_in[1];
    const float* Wk = (const float*)d_in[2];
    const float* Wv = (const float*)d_in[3];
    const float* Wo = (const float*)d_in[4];
    const float* bo = (const float*)d_in[5];
    const float* W1 = (const float*)d_in[6];
    const float* b1 = (const float*)d_in[7];
    const float* W2 = (const float*)d_in[8];
    const float* b2 = (const float*)d_in[9];
    const float* g1 = (const float*)d_in[10];
    const float* g2 = (const float*)d_in[11];
    float* out = (float*)d_out;

    char* ws = (char*)d_ws;
    size_t off = 0;
    auto alloc = [&](size_t bytes) -> void* {
        void* p = ws + off;
        off += (bytes + 255) & ~(size_t)255;
        return p;
    };

    const long long nX   = (long long)BT_ * C_;           // 4M
    const long long nWqkv= (long long)C_ * C_;            // 1M (per matrix)
    const long long nW1  = (long long)C_ * DFF_;          // 4M
    const long long nQKV = (long long)B_ * H_ * T_ * DH_; // 4M
    const long long nFFN = (long long)BT_ * DFF_;         // 16M

    __bf16* xb   = (__bf16*)alloc(nX    * 2);
    __bf16* Wqb  = (__bf16*)alloc(nWqkv * 2);
    __bf16* Wkb  = (__bf16*)alloc(nWqkv * 2);
    __bf16* Wvb  = (__bf16*)alloc(nWqkv * 2);
    __bf16* Wob  = (__bf16*)alloc(nWqkv * 2);
    __bf16* W1b  = (__bf16*)alloc(nW1   * 2);
    __bf16* W2b  = (__bf16*)alloc(nW1   * 2);
    __bf16* qb   = (__bf16*)alloc(nQKV  * 2);
    __bf16* kb   = (__bf16*)alloc(nQKV  * 2);
    __bf16* vTb  = (__bf16*)alloc(nQKV  * 2);
    __bf16* attnb= (__bf16*)alloc(nX    * 2);
    float*  res1 = (float*) alloc(nX    * 4);
    float*  hF   = (float*) alloc(nX    * 4);
    __bf16* hB   = (__bf16*)alloc(nX    * 2);
    __bf16* ffnb = (__bf16*)alloc(nFFN  * 2);
    float*  res2 = (float*) alloc(nX    * 4);
    (void)ws_size; (void)in_sizes; (void)n_in; (void)out_size;

    // 1) convert inputs/weights to bf16
    cvt_f32_bf16<<<2048, 256, 0, stream>>>(x,  xb,  nX);
    cvt_f32_bf16<<<1024, 256, 0, stream>>>(Wq, Wqb, nWqkv);
    cvt_f32_bf16<<<1024, 256, 0, stream>>>(Wk, Wkb, nWqkv);
    cvt_f32_bf16<<<1024, 256, 0, stream>>>(Wv, Wvb, nWqkv);
    cvt_f32_bf16<<<1024, 256, 0, stream>>>(Wo, Wob, nWqkv);
    cvt_f32_bf16<<<2048, 256, 0, stream>>>(W1, W1b, nW1);
    cvt_f32_bf16<<<2048, 256, 0, stream>>>(W2, W2b, nW1);

    // 2) QKV projections (BN=64 == one head per blockIdx.x)
    dim3 gQKV(H_, BT_ / 128);
    gemm_bf16<0><<<gQKV, 256, 0, stream>>>(xb, C_, Wqb, DH_, nullptr, nullptr, nullptr, qb,  C_, 0);
    gemm_bf16<0><<<gQKV, 256, 0, stream>>>(xb, C_, Wkb, DH_, nullptr, nullptr, nullptr, kb,  C_, 0);
    gemm_bf16<1><<<gQKV, 256, 0, stream>>>(xb, C_, Wvb, DH_, nullptr, nullptr, nullptr, vTb, C_, 0);

    // 3) causal flash attention
    attn_kernel<<<B_ * H_ * (T_ / 64), 128, 0, stream>>>(qb, kb, vTb, attnb);

    // 4) output projection + bo + residual(x) -> res1 (f32)
    gemm_bf16<2><<<dim3(C_ / 64, BT_ / 128), 256, 0, stream>>>(
        attnb, C_, Wob, C_, bo, x, res1, nullptr, C_, C_);

    // 5) rmsnorm1 -> hF (f32 residual) + hB (bf16 GEMM input)
    rmsnorm_kernel<<<BT_, 256, 0, stream>>>(res1, g1, hF, hB);

    // 6) FFN1: silu(h @ W1 + b1) -> bf16
    gemm_bf16<3><<<dim3(DFF_ / 64, BT_ / 128), 256, 0, stream>>>(
        hB, C_, W1b, DFF_, b1, nullptr, nullptr, ffnb, C_, DFF_);

    // 7) FFN2: ffn @ W2 + b2 + h residual -> res2 (f32)
    gemm_bf16<2><<<dim3(C_ / 64, BT_ / 128), 256, 0, stream>>>(
        ffnb, DFF_, W2b, C_, b2, hF, res2, nullptr, DFF_, C_);

    // 8) rmsnorm2 -> final output (f32)
    rmsnorm_kernel<<<BT_, 256, 0, stream>>>(res2, g2, out, nullptr);
}
